// FDAFTFineMatching_58368605553074
// MI455X (gfx1250) — matrix operand order repaired and checked by reading, not compiled, not run on hardware
//
#include <hip/hip_runtime.h>
#include <hip/hip_bf16.h>

typedef __attribute__((ext_vector_type(16))) _Float16 v16h;
typedef __attribute__((ext_vector_type(8)))  _Float16 v8h;
typedef __attribute__((ext_vector_type(8)))  float    v8f;

#define PW 72            // LDS pitch for f16 matrices (halves); 144B rows keep 16B alignment
#define PS 65            // LDS pitch for f32 sim matrix (floats); odd-ish stride kills column bank conflicts
#define MW 8192

// ---------------------------------------------------------------------------
// Prep: transpose + convert w_proj / w_fine (64x64 f32, [K][N]) -> f16 [N][K]
// ---------------------------------------------------------------------------
__global__ void fdaft_prep_kernel(const float* __restrict__ wp,
                                  const float* __restrict__ wf,
                                  _Float16* __restrict__ wpT,
                                  _Float16* __restrict__ wfT) {
    int i = blockIdx.x * 256 + threadIdx.x;       // 0..4095
    if (i < 4096) {
        int n = i >> 6, k = i & 63;
        wpT[n * 64 + k] = (_Float16)wp[k * 64 + n];
        wfT[n * 64 + k] = (_Float16)wf[k * 64 + n];
    }
}

// ---------------------------------------------------------------------------
// WMMA operand builders (layouts per CDNA5 ISA 7.12.2, wave32)
// ---------------------------------------------------------------------------
__device__ __forceinline__ v16h load_a16(const _Float16* A, int mBase, int kStep, int lane) {
    const int g = (lane >> 4) & 1;
    const _Float16* p = A + (mBase + (lane & 15)) * PW + kStep + 8 * g;
    v8h lo = *(const v8h*)p;          // K = kStep + 8g + 0..7
    v8h hi = *(const v8h*)(p + 16);   // K = kStep + 8g + 16..23
    return __builtin_shufflevector(lo, hi, 0,1,2,3,4,5,6,7,8,9,10,11,12,13,14,15);
}

// B supplied as B^T stored row-major [N][K] -> contiguous K per lane
__device__ __forceinline__ v16h load_bt16(const _Float16* BT, int nBase, int kStep, int lane) {
    const int g = (lane >> 4) & 1;
    const _Float16* p = BT + (nBase + (lane & 15)) * PW + kStep + 8 * g;
    v8h lo = *(const v8h*)p;          // K = kStep + 8g + 0..7
    v8h hi = *(const v8h*)(p + 16);   // K = kStep + 8g + 16..23
    return __builtin_shufflevector(lo, hi, 0,1,2,3,4,5,6,7,8,9,10,11,12,13,14,15);
}

// 16x16 output tile, K = 64 (two v_wmma_f32_16x16x32_f16)
__device__ __forceinline__ v8f wmma_k64(const _Float16* A, int aRow,
                                        const _Float16* BT, int btRow, int lane) {
    v8f acc = {};
#pragma unroll
    for (int ks = 0; ks < 64; ks += 32) {
        v16h a = load_a16(A, aRow, ks, lane);
        v16h b = load_bt16(BT, btRow, ks, lane);
        acc = __builtin_amdgcn_wmma_f32_16x16x32_f16(false, a, false, b, (short)0, acc,
                                                     false, false);
    }
    return acc;
}

// ---------------------------------------------------------------------------
// Main kernel: one workgroup (8 wave32) per window
// ---------------------------------------------------------------------------
__global__ __launch_bounds__(256)
void fdaft_fine_kernel(const float* __restrict__ f0g, const float* __restrict__ f1g,
                       const int* __restrict__ iic, const int* __restrict__ jic,
                       const _Float16* __restrict__ wpT, const _Float16* __restrict__ wfT,
                       const float* __restrict__ bproj_g, const float* __restrict__ gln_g,
                       const float* __restrict__ bln_g,
                       const float* __restrict__ w1, const float* __restrict__ gm,
                       const float* __restrict__ bm,
                       const float* __restrict__ w2, const float* __restrict__ w3,
                       float* __restrict__ out, int M) {
    __shared__ __align__(16) char smem[62720];
    _Float16* U  = (_Float16*)(smem + 0);        // 128 x PW f16 : proj->GELU output (lives whole kernel)
    _Float16* WT = (_Float16*)(smem + 18432);    // 64 x PW f16 : wpT, then wfT, then misc scratch
    _Float16* Xh = (_Float16*)(smem + 27648);    // 128 x PW f16 : input feats, reused as F (fine output)
    _Float16* F  = Xh;
    float*    S  = (float*)(smem + 46080);       // 64 x PS f32 : sim matrix (phase>=4)
    // S region doubles as small constants before sim is written:
    float* glnS  = (float*)(smem + 46080);       // 64
    float* blnS  = (float*)(smem + 46080 + 256); // 64
    float* bprS  = (float*)(smem + 46080 + 512); // 64
    // WT region reused after fine GEMM:
    char*  mb    = smem + 18432;
    float* rmax  = (float*)(mb + 0);
    float* rsum  = (float*)(mb + 256);
    float* cmax  = (float*)(mb + 512);
    float* csum  = (float*)(mb + 768);
    float* bmax  = (float*)(mb + 1024);          // 256 floats
    int*   bidx  = (int*)(mb + 2048);            // 256 ints
    float* combf = (float*)(mb + 3072);          // 128
    float* bh1   = (float*)(mb + 3584);          // 128
    float* br1   = (float*)(mb + 4096);          // 128
    float* br2   = (float*)(mb + 4608);          // 64
    float* bsub  = (float*)(mb + 4864);          // 4
    float* mst   = (float*)(mb + 4880);          // 2

    const int tid  = threadIdx.x;
    const int w    = tid >> 5;
    const int lane = tid & 31;
    const int m    = blockIdx.x;

    // ---------------- Phase 0: load feats (f32->f16) + wpT + constants ----
    {
        const float* __restrict__ srcs[2] = { f0g + (size_t)m * 4096, f1g + (size_t)m * 4096 };
#pragma unroll
        for (int it = 0; it < 8; ++it) {
            int g4 = tid + 256 * it;             // 0..2047 groups of 4 floats
            int e  = g4 * 4;                     // element 0..8188 (rows 0..127)
            const float4 v = ((const float4*)srcs[e >> 12])[(e & 4095) >> 2];
            _Float16* d = Xh + (e >> 6) * PW + (e & 63);
            d[0] = (_Float16)v.x; d[1] = (_Float16)v.y;
            d[2] = (_Float16)v.z; d[3] = (_Float16)v.w;
        }
        for (int i = tid; i < 4096; i += 256)    // wpT -> WT (pitch PW)
            WT[(i >> 6) * PW + (i & 63)] = wpT[i];
        if (tid < 64)            glnS[tid]       = gln_g[tid];
        else if (tid < 128)      blnS[tid - 64]  = bln_g[tid - 64];
        else if (tid < 192)      bprS[tid - 128] = bproj_g[tid - 128];
    }
    __syncthreads();

    // ---------------- Phase 1: proj GEMM  [128x64] = Xh @ Wp  (+bias) -----
    {
        const int mBase = w * 16;
        const int g = (lane >> 4) & 1;
#pragma unroll
        for (int ct = 0; ct < 4; ++ct) {
            v8f acc = wmma_k64(Xh, mBase, WT, ct * 16, lane);
            const int n = ct * 16 + (lane & 15);
            const float bia = bprS[n];
#pragma unroll
            for (int r = 0; r < 8; ++r)
                U[(mBase + r + 8 * g) * PW + n] = (_Float16)(acc[r] + bia);
        }
    }
    __syncthreads();

    // ---------------- Phase 2: LayerNorm + exact GELU (rows of U) ---------
    //                  threads >=128 concurrently stage wfT into WT
    if (tid < 128) {
        float s = 0.f, ss = 0.f;
        for (int c = 0; c < 64; ++c) { float x = (float)U[tid * PW + c]; s += x; ss += x * x; }
        const float mu = s * (1.f / 64.f);
        const float iv = rsqrtf(ss * (1.f / 64.f) - mu * mu + 1e-5f);
        for (int c = 0; c < 64; ++c) {
            float x = (float)U[tid * PW + c];
            float y = (x - mu) * iv * glnS[c] + blnS[c];
            float ge = 0.5f * y * (1.f + erff(y * 0.70710678118654752f));
            U[tid * PW + c] = (_Float16)ge;
        }
    } else {
        for (int i = tid - 128; i < 4096; i += 128)
            WT[(i >> 6) * PW + (i & 63)] = wfT[i];
    }
    __syncthreads();

    // ---------------- Phase 3: fine GEMM  F = (U @ Wf) / sqrt(64) ---------
    {
        const int mBase = w * 16;
        const int g = (lane >> 4) & 1;
#pragma unroll
        for (int ct = 0; ct < 4; ++ct) {
            v8f acc = wmma_k64(U, mBase, WT, ct * 16, lane);
            const int n = ct * 16 + (lane & 15);
#pragma unroll
            for (int r = 0; r < 8; ++r)
                F[(mBase + r + 8 * g) * PW + n] = (_Float16)(acc[r] * 0.125f);
        }
    }
    __syncthreads();

    // ---------------- Phase 4: sim = f0 @ f1^T / TEMP  (64x64 f32) --------
    {
        const int mBase = (w >> 1) * 16;         // l-tile
        const int g = (lane >> 4) & 1;
#pragma unroll
        for (int cc = 0; cc < 2; ++cc) {
            const int ct = (w & 1) * 2 + cc;     // s-tile
            v8f acc = wmma_k64(F, mBase, F, 64 + ct * 16, lane);
            const int sN = ct * 16 + (lane & 15);
#pragma unroll
            for (int r = 0; r < 8; ++r)
                S[(mBase + r + 8 * g) * PS + sN] = acc[r] * 10.0f;   // /TEMP
        }
    }
    __syncthreads();

    // ---------------- Phase 5: row/col softmax statistics -----------------
    if (tid < 64) {                                       // row l = tid (axis=2)
        float mx = -3.4e38f;
        for (int sc = 0; sc < 64; ++sc) mx = fmaxf(mx, S[tid * PS + sc]);
        float sm = 0.f;
        for (int sc = 0; sc < 64; ++sc) sm += __expf(S[tid * PS + sc] - mx);
        rmax[tid] = mx; rsum[tid] = sm;
    } else if (tid < 128) {                               // col s = tid-64 (axis=1)
        const int sc = tid - 64;
        float mx = -3.4e38f;
        for (int l = 0; l < 64; ++l) mx = fmaxf(mx, S[l * PS + sc]);
        float sm = 0.f;
        for (int l = 0; l < 64; ++l) sm += __expf(S[l * PS + sc] - mx);
        cmax[sc] = mx; csum[sc] = sm;
    }
    __syncthreads();

    // ---------------- Phase 6: conf = smax1*smax2, write out, argmax ------
    float* out_conf = out + (size_t)M * 5 + (size_t)m * 4096;
    {
        float lm = -1.f; int li = 0;
#pragma unroll
        for (int it = 0; it < 16; ++it) {
            const int e = tid + 256 * it;                 // flat l*64+s, coalesced
            const int l = e >> 6, sc = e & 63;
            const float x = S[l * PS + sc];
            const float v = (__expf(x - cmax[sc]) / csum[sc]) *
                            (__expf(x - rmax[l]) / rsum[l]);
            out_conf[e] = v;
            if (v > lm || (v == lm && e < li)) { lm = v; li = e; }
        }
        bmax[tid] = lm; bidx[tid] = li;
    }
    __syncthreads();
    for (int off = 128; off > 0; off >>= 1) {
        if (tid < off) {
            float vo = bmax[tid + off]; int io = bidx[tid + off];
            if (vo > bmax[tid] || (vo == bmax[tid] && io < bidx[tid])) {
                bmax[tid] = vo; bidx[tid] = io;
            }
        }
        __syncthreads();
    }

    // ---------------- Phase 7: subpixel MLP on matched descriptors --------
    const int idx = bidx[0];
    const int ii = idx >> 6, jj = idx & 63;
    if (tid < 64)        combf[tid] = (float)U[ii * PW + tid];
    else if (tid < 128)  combf[tid] = (float)U[(64 + jj) * PW + (tid - 64)];
    __syncthreads();
    if (tid < 128) {                                      // h1 = comb @ w_mlp1
        float a = 0.f;
        for (int k = 0; k < 128; ++k) a += combf[k] * w1[k * 128 + tid];
        bh1[tid] = a;
    }
    __syncthreads();
    if (tid == 0) {
        float s = 0.f, ss = 0.f;
        for (int k = 0; k < 128; ++k) { float v = bh1[k]; s += v; ss += v * v; }
        float mu = s * (1.f / 128.f);
        mst[0] = mu; mst[1] = ss * (1.f / 128.f) - mu * mu;
    }
    __syncthreads();
    if (tid < 128) {                                      // ReLU(LayerNorm(h1))
        float y = (bh1[tid] - mst[0]) * rsqrtf(mst[1] + 1e-5f) * gm[tid] + bm[tid];
        br1[tid] = fmaxf(y, 0.f);
    }
    __syncthreads();
    if (tid < 64) {                                       // ReLU(h @ w_mlp2)
        float a = 0.f;
        for (int k = 0; k < 128; ++k) a += br1[k] * w2[k * 64 + tid];
        br2[tid] = fmaxf(a, 0.f);
    }
    __syncthreads();
    if (tid < 4) {                                        // sub = h @ w_mlp3
        float a = 0.f;
        for (int k = 0; k < 64; ++k) a += br2[k] * w3[k * 4 + tid];
        bsub[tid] = a;
    }
    __syncthreads();

    // ---------------- Phase 8: keypoint coordinates + mconf ---------------
    if (tid < 4) {
        const int ic = iic[m], jc = jic[m];
        float base;
        if (tid == 0)      base = (float)((ii & 7)  + (ic % 104) * 2);
        else if (tid == 1) base = (float)((ii >> 3) + (ic / 104) * 2);
        else if (tid == 2) base = (float)((jj & 7)  + (jc % 104) * 2);
        else               base = (float)((jj >> 3) + (jc / 104) * 2);
        out[(size_t)m * 4 + tid] = (base + tanhf(bsub[tid]) * 0.5f) * 4.0f;
    }
    if (tid == 0) {
        const float mm = bmax[0];
        out[(size_t)M * 4 + m] = (mm > 8e-5f) ? mm : 0.0f;   // FINE_THR
    }
}

// ---------------------------------------------------------------------------
extern "C" void kernel_launch(void* const* d_in, const int* in_sizes, int n_in,
                              void* d_out, int out_size, void* d_ws, size_t ws_size,
                              hipStream_t stream) {
    const float* f0  = (const float*)d_in[0];
    const float* f1  = (const float*)d_in[1];
    const int*   iic = (const int*)d_in[2];
    const int*   jic = (const int*)d_in[3];
    // d_in[4] = b_ids_c (unused by reference)
    const float* wp  = (const float*)d_in[5];
    const float* bp  = (const float*)d_in[6];
    const float* gl  = (const float*)d_in[7];
    const float* bl  = (const float*)d_in[8];
    const float* wf  = (const float*)d_in[9];
    const float* w1  = (const float*)d_in[10];
    const float* gmp = (const float*)d_in[11];
    const float* bmp = (const float*)d_in[12];
    const float* w2  = (const float*)d_in[13];
    const float* w3  = (const float*)d_in[14];
    const int M = in_sizes[2];

    _Float16* wpT = (_Float16*)d_ws;
    _Float16* wfT = wpT + 4096;

    fdaft_prep_kernel<<<16, 256, 0, stream>>>(wp, wf, wpT, wfT);
    fdaft_fine_kernel<<<M, 256, 0, stream>>>(f0, f1, iic, jic, wpT, wfT,
                                             bp, gl, bl, w1, gmp, bmp, w2, w3,
                                             (float*)d_out, M);
}
